// SRNN_GLN_Cell_18494129176993
// MI455X (gfx1250) — matrix operand order repaired and checked by reading, not compiled, not run on hardware
//
#include <hip/hip_runtime.h>
#include <hip/hip_bf16.h>

// ---------------------------------------------------------------------------
// SRNN GLN cell for MI455X (gfx1250).
//   Kernel 1: fused 2x gated-linear GEMM (bf16 WMMA, fp32 accum), weights
//             staged in LDS (320KB/WGP), h kept on-chip, b -> d_out.
//             Fragment loads are contiguous per lane -> ds_load_b128 pairs.
//   Kernel 2: sequential scan, rotation-of-coordinates removes the roll()
//             lane shuffle entirely; runs in place over d_out.
// ---------------------------------------------------------------------------

#define Dm   256
#define MB   64          // rows per block tile of the flattened [B*T, D] GEMM
#define TSEQ 4096
#define NB   16          // batch

typedef __attribute__((ext_vector_type(16))) __bf16        v16bf;
typedef __attribute__((ext_vector_type(8)))  float         v8f;

struct U8frag { uint4 a, b; };   // 32 bytes = one 16x16x32 bf16 operand fragment

__device__ __forceinline__ unsigned short f2bf(float f) {
    unsigned u = __builtin_bit_cast(unsigned, f);
    unsigned r = (u + 0x7FFFu + ((u >> 16) & 1u)) >> 16;   // round-to-nearest-even
    return (unsigned short)r;
}
__device__ __forceinline__ unsigned int pk2(float lo, float hi) {
    return (unsigned)f2bf(lo) | ((unsigned)f2bf(hi) << 16);
}

// A fragment: 16x32 bf16 (ISA 16-bit A layout). Per lane this is two
// contiguous 16B runs: K = kb+lhi*8+{0..7} and +{16..23}  -> 2x ds_load_b128.
__device__ __forceinline__ v16bf load_afrag(const unsigned short* tile, int m0,
                                            int kb, int lan15, int lhi) {
    const unsigned short* rp = tile + (m0 + lan15) * Dm + kb + lhi * 8;
    U8frag r;
    r.a = *(const uint4*)(rp);        // K pairs {0,1..6,7}
    r.b = *(const uint4*)(rp + 16);   // K pairs {16,17..22,23}
    return __builtin_bit_cast(v16bf, r);
}

// B fragment: 32x16 bf16 = W^T tile; B[k][n] = W[n][k]. Per lane this is two
// contiguous 16B runs W[n][kb+lhi*8+{0..7}] and +{16..23} -> 2x ds_load_b128.
__device__ __forceinline__ v16bf load_bfrag(const unsigned short* w, int nb,
                                            int kb, int lan15, int lhi) {
    const unsigned short* rp = w + (nb + lan15) * Dm + kb + lhi * 8;
    U8frag r;
    r.a = *(const uint4*)(rp);
    r.b = *(const uint4*)(rp + 16);
    return __builtin_bit_cast(v16bf, r);
}

// Cooperative fp32->bf16 staging of a weight-matrix pair into LDS.
__device__ __forceinline__ void stage_w(const float* __restrict__ Wa,
                                        const float* __restrict__ Wb,
                                        unsigned short* wl, int tid) {
    for (int idx = tid * 4; idx < 2 * Dm * Dm; idx += 256 * 4) {
        const float* s = (idx < Dm * Dm) ? (Wa + idx) : (Wb + (idx - Dm * Dm));
        float4 v = *(const float4*)s;
        uint2 p; p.x = pk2(v.x, v.y); p.y = pk2(v.z, v.w);
        *(uint2*)(wl + idx) = p;
    }
}

// One gated-linear stage for a 64x256 tile:  act = (A@Wa^T+ba)*sigmoid(A@Wb^T+bb)
// Wave w owns M-strip (w>>1)*16 and N-half (w&1)*128: 8 N-frags x 2 paths.
template <bool TO_GLOBAL>
__device__ __forceinline__ void gln_stage(const unsigned short* atile,
                                          const unsigned short* wl,
                                          const float* __restrict__ biasA,
                                          const float* __restrict__ biasB,
                                          unsigned short* hout,
                                          float* __restrict__ gout, int row0) {
    const int lane  = threadIdx.x & 31;
    const int wave  = threadIdx.x >> 5;
    const int mb    = (wave >> 1) * 16;
    const int nh    = (wave & 1) * 128;
    const int lan15 = lane & 15;
    const int lhi   = lane >> 4;
    const unsigned short* w0l = wl;
    const unsigned short* w1l = wl + Dm * Dm;

    v8f acc0[8], acc1[8];
#pragma unroll
    for (int i = 0; i < 8; ++i) {
        acc0[i] = (v8f){0.f, 0.f, 0.f, 0.f, 0.f, 0.f, 0.f, 0.f};
        acc1[i] = (v8f){0.f, 0.f, 0.f, 0.f, 0.f, 0.f, 0.f, 0.f};
    }
#pragma unroll
    for (int k8 = 0; k8 < 8; ++k8) {               // K = 256 in steps of 32
        const int kb = k8 * 32;
        v16bf a = load_afrag(atile, mb, kb, lan15, lhi);
#pragma unroll
        for (int nf = 0; nf < 8; ++nf) {
            const int nb = nh + nf * 16;
            v16bf bA = load_bfrag(w0l, nb, kb, lan15, lhi);
            v16bf bB = load_bfrag(w1l, nb, kb, lan15, lhi);
            acc0[nf] = __builtin_amdgcn_wmma_f32_16x16x32_bf16(
                false, a, false, bA, (short)0, acc0[nf], false, false);
            acc1[nf] = __builtin_amdgcn_wmma_f32_16x16x32_bf16(
                false, a, false, bB, (short)0, acc1[nf], false, false);
        }
    }
    // Epilogue: bias + gate. C layout: lane<16 -> N=lane, M=i; lane>=16 -> M=i+8.
#pragma unroll
    for (int nf = 0; nf < 8; ++nf) {
        const int n  = nh + nf * 16 + lan15;
        const float bA = biasA[n];
        const float bB = biasB[n];
#pragma unroll
        for (int i = 0; i < 8; ++i) {
            float y0 = acc0[nf][i] + bA;
            float y1 = acc1[nf][i] + bB;
            float g  = y0 * (1.f / (1.f + __expf(-y1)));
            int   m  = mb + lhi * 8 + i;
            if (TO_GLOBAL)
                gout[(size_t)(row0 + m) * Dm + n] = g;
            else
                hout[m * Dm + n] = f2bf(g);
        }
    }
}

__global__ void __launch_bounds__(256, 1)
gln_kernel(const float* __restrict__ x,
           const float* __restrict__ W0, const float* __restrict__ b0,
           const float* __restrict__ W1, const float* __restrict__ b1,
           const float* __restrict__ W2, const float* __restrict__ b2,
           const float* __restrict__ W3, const float* __restrict__ b3,
           float* __restrict__ out) {
    extern __shared__ unsigned char smem[];
    unsigned short* wl = (unsigned short*)smem;                     // 256 KB: W pair, bf16
    unsigned short* xb = (unsigned short*)(smem + 262144);          //  32 KB: x tile, bf16
    unsigned short* hb = (unsigned short*)(smem + 262144 + 32768);  //  32 KB: h tile, bf16

    const int tid  = threadIdx.x;
    const int row0 = blockIdx.x * MB;

    // Stage x tile (fp32 -> bf16)
    {
        const float* src = x + (size_t)row0 * Dm;
        for (int idx = tid * 4; idx < MB * Dm; idx += 256 * 4) {
            float4 v = *(const float4*)(src + idx);
            uint2 p; p.x = pk2(v.x, v.y); p.y = pk2(v.z, v.w);
            *(uint2*)(xb + idx) = p;
        }
    }
    stage_w(W0, W1, wl, tid);
    __syncthreads();

    gln_stage<false>(xb, wl, b0, b1, hb, nullptr, row0);   // h -> LDS
    __syncthreads();                                       // all waves done with W0/W1

    stage_w(W2, W3, wl, tid);
    __syncthreads();

    gln_stage<true>(hb, wl, b2, b3, nullptr, out, row0);   // b -> d_out (fp32)
}

// Scan: u[t][j] = relu(b[t][(j+t)&255] + u[t-1][j]); pure elementwise in j.
// In-place on d_out (each (t,d) cell read+written by exactly one thread).
__global__ void __launch_bounds__(256, 1)
scan_kernel(const float* __restrict__ hidden, float* __restrict__ out,
            float* __restrict__ last) {
    const int b = blockIdx.x;
    const int j = threadIdx.x;
    float* p = out + (size_t)b * TSEQ * Dm;

    float u = hidden[b * Dm + ((j + 255) & 255)];          // roll(hidden,1)[j]
#pragma unroll 2
    for (int t = 0; t < TSEQ; t += 4) {
        float* q0 = p + (size_t)(t + 0) * Dm + ((j + t + 0) & 255);
        float* q1 = p + (size_t)(t + 1) * Dm + ((j + t + 1) & 255);
        float* q2 = p + (size_t)(t + 2) * Dm + ((j + t + 2) & 255);
        float* q3 = p + (size_t)(t + 3) * Dm + ((j + t + 3) & 255);
        float c0 = *q0, c1 = *q1, c2 = *q2, c3 = *q3;      // off critical path
        if (t + 8 < TSEQ)
            __builtin_prefetch(p + (size_t)(t + 8) * Dm + j, 0, 0);
        u = fmaxf(c0 + u, 0.f); *q0 = u;
        u = fmaxf(c1 + u, 0.f); *q1 = u;
        u = fmaxf(c2 + u, 0.f); *q2 = u;
        u = fmaxf(c3 + u, 0.f); *q3 = u;
    }
    last[b * Dm + ((j + 255) & 255)] = u;                  // out[T-1] in u-coords
}

extern "C" void kernel_launch(void* const* d_in, const int* in_sizes, int n_in,
                              void* d_out, int out_size, void* d_ws, size_t ws_size,
                              hipStream_t stream) {
    (void)in_sizes; (void)n_in; (void)out_size; (void)d_ws; (void)ws_size;
    const float* x      = (const float*)d_in[0];
    const float* hidden = (const float*)d_in[1];
    const float* W0 = (const float*)d_in[2]; const float* b0 = (const float*)d_in[3];
    const float* W1 = (const float*)d_in[4]; const float* b1 = (const float*)d_in[5];
    const float* W2 = (const float*)d_in[6]; const float* b2 = (const float*)d_in[7];
    const float* W3 = (const float*)d_in[8]; const float* b3 = (const float*)d_in[9];

    float* out  = (float*)d_out;
    float* last = out + (size_t)NB * TSEQ * Dm;

    const size_t lds = 262144 + 32768 + 32768;   // 320 KB (full WGP LDS)
    hipFuncSetAttribute(reinterpret_cast<const void*>(gln_kernel),
                        hipFuncAttributeMaxDynamicSharedMemorySize, (int)lds);

    const int n_tiles = (NB * TSEQ) / MB;        // 1024 blocks of 64 rows
    gln_kernel<<<n_tiles, 256, lds, stream>>>(x, W0, b0, W1, b1, W2, b2, W3, b3, out);
    scan_kernel<<<NB, 256, 0, stream>>>(hidden, out, last);
}
